// GPTAve_80504866996430
// MI455X (gfx1250) — compile-verified, matrix operand order checked by reading, not run on hardware
//
#include <hip/hip_runtime.h>
#include <math.h>

typedef __attribute__((ext_vector_type(16))) _Float16 v16h;
typedef __attribute__((ext_vector_type(8)))  _Float16 v8h;
typedef __attribute__((ext_vector_type(8)))  float    v8f;

static constexpr int   S_   = 2048;
static constexpr int   D_   = 512;
static constexpr int   F_   = 512;
static constexpr int   L_   = 4;
static constexpr int   V_   = 32000;
static constexpr float EPS_ = 1e-5f;

union HFrag { v16h v; v8h h[2]; };

// ---------------------------------------------------------------------------
// f16 GEMM, all operands pre-converted:
//   A  : f16 [M,K] row-major
//   Bt : f16 [N,K] row-major  (i.e. B transposed -> contiguous K per column)
// Epilogue: v = alpha*acc (+ bias[col]) (causal mask), then one of:
//   outmode 0: Cf[row*N+col] = v          (f32)
//   outmode 1: Ch[row*N+col] = (f16)v
//   outmode 2: Ch[col*M+row] = (f16)v     (f16, transposed output)
// One wave computes a 32x64 tile: 8 v_wmma_f32_16x16x32_f16 per K=32 step
// fed by 12 contiguous 16-byte loads. Block = 128 threads = 4 waves along M.
// ---------------------------------------------------------------------------
__global__ __launch_bounds__(128)
void gemm_wmma(const _Float16* __restrict__ A, const _Float16* __restrict__ Bt,
               const float* __restrict__ bias,
               float* __restrict__ Cf, _Float16* __restrict__ Ch,
               int M, int N, int K, float alpha, int causal, int outmode)
{
    const int lane   = threadIdx.x & 31;
    const int wave   = threadIdx.x >> 5;
    const int m0     = (blockIdx.y * 4 + wave) * 32;   // 32-row wave tile
    const int n_base = blockIdx.x * 64;                // 64-col wave tile
    if (m0 >= M) return;                               // wave-uniform (EXEC all-1s)

    const int half_sel = lane >> 4;                    // K-half select per ISA layout
    const int koff     = half_sel * 8;
    const int ncol     = lane & 15;
    const int mrow     = lane & 15;

    v8f acc[2][4];
    #pragma unroll
    for (int mi = 0; mi < 2; ++mi)
        #pragma unroll
        for (int t = 0; t < 4; ++t)
            #pragma unroll
            for (int r = 0; r < 8; ++r) acc[mi][t][r] = 0.0f;

    const _Float16* arow0 = A + (size_t)(m0 + mrow) * K;
    const _Float16* arow1 = arow0 + (size_t)16 * K;

    for (int k0 = 0; k0 < K; k0 += 32) {
        __builtin_prefetch(arow0 + k0 + 32, 0, 3);     // global_prefetch_b8
        HFrag a[2];
        a[0].h[0] = *(const v8h*)(arow0 + k0 + koff);
        a[0].h[1] = *(const v8h*)(arow0 + k0 + 16 + koff);
        a[1].h[0] = *(const v8h*)(arow1 + k0 + koff);
        a[1].h[1] = *(const v8h*)(arow1 + k0 + 16 + koff);

        #pragma unroll
        for (int t = 0; t < 4; ++t) {
            const _Float16* brow = Bt + (size_t)(n_base + t * 16 + ncol) * K + k0;
            HFrag b;
            b.h[0] = *(const v8h*)(brow + koff);
            b.h[1] = *(const v8h*)(brow + 16 + koff);
            acc[0][t] = __builtin_amdgcn_wmma_f32_16x16x32_f16(
                false, a[0].v, false, b.v, (short)0, acc[0][t], false, false);
            acc[1][t] = __builtin_amdgcn_wmma_f32_16x16x32_f16(
                false, a[1].v, false, b.v, (short)0, acc[1][t], false, false);
        }
    }

    // epilogue: C/D layout -> row = m0 + 16*mi + r + 8*half_sel, col = n_base+16*t+ncol
    #pragma unroll
    for (int mi = 0; mi < 2; ++mi) {
        #pragma unroll
        for (int t = 0; t < 4; ++t) {
            const int col = n_base + t * 16 + ncol;
            #pragma unroll
            for (int r = 0; r < 8; ++r) {
                const int row = m0 + mi * 16 + r + half_sel * 8;
                float v = acc[mi][t][r] * alpha;
                if (bias) v += bias[col];
                if (causal && col > row) v = -INFINITY;
                if (outmode == 0)      Cf[(size_t)row * N + col] = v;
                else if (outmode == 1) Ch[(size_t)row * N + col] = (_Float16)v;
                else                   Ch[(size_t)col * M + row] = (_Float16)v;
            }
        }
    }
}

// ---------------------------------------------------------------------------
// LDS-tiled transpose + convert: src f32 [R,C] -> dst f16 [C,R]
// ---------------------------------------------------------------------------
__global__ __launch_bounds__(256)
void transpose_f32_to_f16(const float* __restrict__ src, _Float16* __restrict__ dst,
                          int R, int C)
{
    __shared__ _Float16 tile[32][33];
    const int tx = threadIdx.x & 31;
    const int ty = threadIdx.x >> 5;           // 0..7
    const int c0 = blockIdx.x * 32;
    const int r0 = blockIdx.y * 32;
    #pragma unroll
    for (int i = 0; i < 32; i += 8)
        tile[ty + i][tx] = (_Float16)src[(size_t)(r0 + ty + i) * C + (c0 + tx)];
    __syncthreads();
    #pragma unroll
    for (int i = 0; i < 32; i += 8)
        dst[(size_t)(c0 + ty + i) * R + (r0 + tx)] = tile[tx][ty + i];
}

// elementwise f32 -> f16
__global__ __launch_bounds__(256)
void f32_to_f16(const float* __restrict__ src, _Float16* __restrict__ dst, int n)
{
    const int i = blockIdx.x * 256 + threadIdx.x;
    if (i < n) dst[i] = (_Float16)src[i];
}

// ---------------------------------------------------------------------------
// Row softmax. If Y != nullptr, write f16 probabilities to Y (X untouched);
// else normalize X in place (f32). Handles -inf masked entries and N=32000.
// ---------------------------------------------------------------------------
__global__ __launch_bounds__(256)
void softmax_rows(float* __restrict__ X, _Float16* __restrict__ Y, int N)
{
    const int tid = threadIdx.x;
    float* x = X + (size_t)blockIdx.x * N;
    _Float16* y = Y ? Y + (size_t)blockIdx.x * N : nullptr;
    __shared__ float red[256];

    float m = -INFINITY;
    for (int i = tid; i < N; i += 256) m = fmaxf(m, x[i]);
    red[tid] = m; __syncthreads();
    for (int o = 128; o > 0; o >>= 1) {
        if (tid < o) red[tid] = fmaxf(red[tid], red[tid + o]);
        __syncthreads();
    }
    m = red[0]; __syncthreads();

    float s = 0.0f;
    for (int i = tid; i < N; i += 256) s += __expf(x[i] - m);
    red[tid] = s; __syncthreads();
    for (int o = 128; o > 0; o >>= 1) {
        if (tid < o) red[tid] += red[tid + o];
        __syncthreads();
    }
    const float inv = 1.0f / red[0];

    for (int i = tid; i < N; i += 256) {
        const float v = __expf(x[i] - m) * inv;
        if (y) y[i] = (_Float16)v;
        else   x[i] = v;
    }
}

// ---------------------------------------------------------------------------
// LayerNorm(Z + R): writes f32 (Of) and f16 (Oh) copies (biased variance).
// ---------------------------------------------------------------------------
__global__ __launch_bounds__(256)
void add_layernorm(const float* __restrict__ Z, const float* __restrict__ R,
                   const float* __restrict__ g, const float* __restrict__ b,
                   float* __restrict__ Of, _Float16* __restrict__ Oh, int N)
{
    const int tid = threadIdx.x;
    const float* z = Z + (size_t)blockIdx.x * N;
    const float* r = R + (size_t)blockIdx.x * N;
    float*       o = Of + (size_t)blockIdx.x * N;
    _Float16*    oh = Oh + (size_t)blockIdx.x * N;
    __shared__ float s1[256], s2[256];

    float acc = 0.0f, accsq = 0.0f;
    for (int i = tid; i < N; i += 256) {
        const float v = z[i] + r[i];
        o[i] = v;
        acc += v; accsq += v * v;
    }
    s1[tid] = acc; s2[tid] = accsq; __syncthreads();
    for (int off = 128; off > 0; off >>= 1) {
        if (tid < off) { s1[tid] += s1[tid + off]; s2[tid] += s2[tid + off]; }
        __syncthreads();
    }
    const float mu  = s1[0] / (float)N;
    const float var = s2[0] / (float)N - mu * mu;
    const float inv = rsqrtf(var + EPS_);
    __syncthreads();
    for (int i = tid; i < N; i += 256) {
        const float v = (o[i] - mu) * inv * g[i] + b[i];
        o[i]  = v;
        oh[i] = (_Float16)v;
    }
}

// ---------------------------------------------------------------------------
extern "C" void kernel_launch(void* const* d_in, const int* in_sizes, int n_in,
                              void* d_out, int out_size, void* d_ws, size_t ws_size,
                              hipStream_t stream)
{
    (void)in_sizes; (void)n_in; (void)out_size; (void)ws_size;

    const float* x_in = (const float*)d_in[0];
    const float* Wq = (const float*)d_in[1];
    const float* bq = (const float*)d_in[2];
    const float* Wk = (const float*)d_in[3];
    const float* bk = (const float*)d_in[4];
    const float* Wv = (const float*)d_in[5];
    const float* bv = (const float*)d_in[6];
    const float* Wp = (const float*)d_in[7];
    const float* bp = (const float*)d_in[8];
    const float* g1 = (const float*)d_in[9];
    const float* be1 = (const float*)d_in[10];
    const float* W1 = (const float*)d_in[11];
    const float* b1 = (const float*)d_in[12];
    const float* W2 = (const float*)d_in[13];
    const float* b2 = (const float*)d_in[14];
    const float* g2 = (const float*)d_in[15];
    const float* be2 = (const float*)d_in[16];
    const float* Wf = (const float*)d_in[17];
    const float* bf = (const float*)d_in[18];
    float* out = (float*)d_out;

    // ---- workspace carve-up (all sizes are multiples of 64 elements) ----
    char* w = (char*)d_ws;
    auto allocF = [&](size_t n) { float* p = (float*)w; w += n * sizeof(float); return p; };
    auto allocH = [&](size_t n) { _Float16* p = (_Float16*)w; w += n * sizeof(_Float16); return p; };

    float* xa  = allocF((size_t)S_ * D_);     // f32 activation (post-layer)
    float* xb  = allocF((size_t)S_ * D_);     // f32 activation (mid-layer)
    float* sc  = allocF((size_t)S_ * S_);     // raw scores f32
    float* zf  = allocF((size_t)S_ * F_);     // attn proj output f32
    float* h2  = allocF((size_t)S_ * D_);     // FFN output f32

    _Float16* x16   = allocH((size_t)S_ * D_);
    _Float16* xb16  = allocH((size_t)S_ * D_);
    _Float16* q16   = allocH((size_t)S_ * F_);
    _Float16* k16   = allocH((size_t)S_ * F_);
    _Float16* vT16  = allocH((size_t)S_ * F_);    // [F,S]
    _Float16* att16 = allocH((size_t)S_ * S_);
    _Float16* z116  = allocH((size_t)S_ * F_);
    _Float16* h16   = allocH((size_t)S_ * 4 * F_);
    _Float16* WqT   = allocH((size_t)D_ * F_);
    _Float16* WkT   = allocH((size_t)D_ * F_);
    _Float16* WvT   = allocH((size_t)D_ * F_);
    _Float16* WpT   = allocH((size_t)F_ * F_);
    _Float16* W1T   = allocH((size_t)F_ * 4 * F_);
    _Float16* W2T   = allocH((size_t)4 * F_ * D_);
    _Float16* WfT   = allocH((size_t)D_ * V_);

    const float scale = 1.0f / sqrtf((float)F_);

    auto gemm = [&](const _Float16* A, const _Float16* Bt, const float* bias,
                    float* Cf, _Float16* Ch, int M, int N, int K,
                    float alpha, int causal, int outmode) {
        dim3 grid(N / 64, M / 128);
        gemm_wmma<<<grid, 128, 0, stream>>>(A, Bt, bias, Cf, Ch, M, N, K,
                                            alpha, causal, outmode);
    };
    auto transp = [&](const float* src, _Float16* dst, int R, int C) {
        dim3 grid(C / 32, R / 32);
        transpose_f32_to_f16<<<grid, 256, 0, stream>>>(src, dst, R, C);
    };

    // x16 = f16(x_in)
    f32_to_f16<<<(S_ * D_) / 256, 256, 0, stream>>>(x_in, x16, S_ * D_);

    const float* xres = x_in;   // f32 residual source for first LN of each layer
    for (int l = 0; l < L_; ++l) {
        const size_t wDF = (size_t)l * D_ * F_;
        const size_t wFF = (size_t)l * F_ * F_;
        const size_t w1o = (size_t)l * F_ * 4 * F_;
        const size_t w2o = (size_t)l * 4 * F_ * D_;

        // weight transposes (f32 [K,N] -> f16 [N,K])
        transp(Wq + wDF, WqT, D_, F_);
        transp(Wk + wDF, WkT, D_, F_);
        transp(Wv + wDF, WvT, D_, F_);
        transp(Wp + wFF, WpT, F_, F_);
        transp(W1 + w1o, W1T, F_, 4 * F_);
        transp(W2 + w2o, W2T, 4 * F_, D_);

        // q, k (f16 out); v (f16 transposed out [F,S])
        gemm(x16, WqT, bq + (size_t)l * F_, nullptr, q16, S_, F_, D_, 1.0f, 0, 1);
        gemm(x16, WkT, bk + (size_t)l * F_, nullptr, k16, S_, F_, D_, 1.0f, 0, 1);
        gemm(x16, WvT, bv + (size_t)l * F_, nullptr, vT16, S_, F_, D_, 1.0f, 0, 2);
        // scores = (q @ k^T)*scale + causal  (k16 [S,F] is already the Bt layout)
        gemm(q16, k16, nullptr, sc, nullptr, S_, S_, F_, scale, 1, 0);
        softmax_rows<<<S_, 256, 0, stream>>>(sc, att16, S_);
        // z1 = att @ v        (Bt = vT16 [F,S])
        gemm(att16, vT16, nullptr, nullptr, z116, S_, F_, S_, 1.0f, 0, 1);
        // z = z1 @ Wp + bp  (f32)
        gemm(z116, WpT, bp + (size_t)l * F_, zf, nullptr, S_, F_, F_, 1.0f, 0, 0);
        // x_mid = LN(z + res)
        add_layernorm<<<S_, 256, 0, stream>>>(zf, xres, g1 + (size_t)l * F_,
                                              be1 + (size_t)l * F_, xb, xb16, F_);
        // FFN (no activation in reference)
        gemm(xb16, W1T, b1 + (size_t)l * 4 * F_, nullptr, h16, S_, 4 * F_, F_, 1.0f, 0, 1);
        gemm(h16, W2T, b2 + (size_t)l * D_, h2, nullptr, S_, D_, 4 * F_, 1.0f, 0, 0);
        // x = LN(h2 + x_mid) -> xa (f32) + x16 (f16 for next layer / logits)
        add_layernorm<<<S_, 256, 0, stream>>>(h2, xb, g2 + (size_t)l * D_,
                                              be2 + (size_t)l * D_, xa, x16, D_);
        xres = xa;
    }

    // logits = x @ Wf + bf -> d_out (f32), softmax rows over V
    transp(Wf, WfT, D_, V_);
    gemm(x16, WfT, bf, out, nullptr, S_, V_, D_, 1.0f, 0, 0);
    softmax_rows<<<S_, 256, 0, stream>>>(out, nullptr, V_);
}
